// MFCCExtractor_1271310319837
// MI455X (gfx1250) — compile-verified, matrix operand order checked by reading, not run on hardware
//
#include <hip/hip_runtime.h>
#include <math.h>

#ifndef M_PI
#define M_PI 3.14159265358979323846
#endif

typedef __attribute__((ext_vector_type(16))) _Float16 v16h;
typedef __attribute__((ext_vector_type(8)))  float    v8f;

// ---------------- problem constants ----------------
#define SAMPLES   320000
#define CHANNELS  64
#define WIN       640
#define STEP      320
#define NFFT      1024
#define NBINS     513
#define NT1       34            // stage-1 N tiles (544 = 34*16, pads 513 bins)
#define KS1       20            // stage-1 K steps (640 = 20*32)
#define NMEL      40
#define NT2       3             // stage-2 N tiles (48 cols)
#define KS2       17            // stage-2 K steps (544 = 17*32)
#define NMFCC     13
#define KS3       2             // stage-3 K steps (64 = 2*32, pads 40 mel)
#define NFRAMES   999
#define FBLOCKS   63            // ceil(999/16)
#define CLIPV     10.0f

// workspace offsets in _Float16 elements
#define N_COS     (NT1*KS1*32*16)   // 348160
#define OFF_COS   0
#define OFF_SIN   (OFF_COS + N_COS)
#define N_MELW    (NT2*KS2*32*16)   // 26112
#define OFF_MEL   (OFF_SIN + N_COS)
#define N_DCTW    (KS3*32*16)       // 1024
#define OFF_DCT   (OFF_MEL + N_MELW)

// K-index pattern inside a 16-bit WMMA fragment (A and B operands):
// element j of lane L covers K = 16*(j>>3) + 8*(L>>4) + (j&7)  (per ISA 7.12.2)
__device__ __forceinline__ int kpat(int lane, int j) {
    return ((j >> 3) << 4) + ((lane >> 4) << 3) + (j & 7);
}

// ---------------- init: Hann-folded DFT cos/sin in B-fragment layout ----------------
__global__ void build_dft(_Float16* __restrict__ ws) {
    int idx = blockIdx.x * blockDim.x + threadIdx.x;
    if (idx >= N_COS) return;
    int nt = idx / (KS1 * 512);
    int ks = (idx / 512) % KS1;
    int ln = (idx >> 4) & 31;
    int j  = idx & 15;
    int n   = ks * 32 + kpat(ln, j);       // time sample 0..639
    int bin = nt * 16 + (ln & 15);         // frequency bin 0..543
    float c = 0.f, s = 0.f;
    if (bin < NBINS) {
        double h = 0.5 - 0.5 * cos(2.0 * M_PI * (double)n / (double)WIN);
        int r = (bin * n) & (NFFT - 1);    // exact phase mod 1024
        double ang = 2.0 * M_PI * (double)r / (double)NFFT;
        c = (float)(h * cos(ang));
        s = (float)(h * sin(ang));         // sign irrelevant: Im is squared
    }
    ws[OFF_COS + idx] = (_Float16)c;
    ws[OFF_SIN + idx] = (_Float16)s;
}

// ---------------- init: mel filterbank (TF MfccMelFilterbank) ----------------
__global__ void build_mel(_Float16* __restrict__ ws) {
    int idx = blockIdx.x * blockDim.x + threadIdx.x;
    if (idx >= N_MELW) return;
    int nt = idx / (KS2 * 512);
    int ks = (idx / 512) % KS2;
    int ln = (idx >> 4) & 31;
    int j  = idx & 15;
    int i  = ks * 32 + kpat(ln, j);        // freq bin 0..543
    int mc = nt * 16 + (ln & 15);          // mel channel 0..47
    float val = 0.f;
    // hz_per_bin = 0.5*16000/512 = 15.625 ; start_index = 2 ; end_index = 486
    if (mc < NMEL && i >= 2 && i <= 486) {
        double mel_lo  = 1127.0 * log1p(20.0 / 700.0);
        double mel_hi  = 1127.0 * log1p(7600.0 / 700.0);
        double spacing = (mel_hi - mel_lo) / (double)(NMEL + 1);
        double melf    = 1127.0 * log1p((double)i * 15.625 / 700.0);
        int chn = 0;
        while (chn < NMEL && (mel_lo + spacing * (chn + 1)) < melf) ++chn;
        --chn;                             // may be -1
        double w;
        if (chn >= 0) w = (mel_lo + spacing * (chn + 2) - melf) / spacing;
        else          w = (mel_lo + spacing - melf) / spacing;
        if (mc == chn)          val = (float)w;
        else if (mc == chn + 1) val = (float)(1.0 - w);
    }
    ws[OFF_MEL + idx] = (_Float16)val;
}

// ---------------- init: DCT-II matrix ----------------
__global__ void build_dct(_Float16* __restrict__ ws) {
    int idx = blockIdx.x * blockDim.x + threadIdx.x;
    if (idx >= N_DCTW) return;
    int ks = idx / 512;
    int ln = (idx >> 4) & 31;
    int j  = idx & 15;
    int K  = ks * 32 + kpat(ln, j);        // mel bin 0..63
    int N  = ln & 15;                      // mfcc index 0..15
    float val = 0.f;
    if (K < NMEL && N < NMFCC)
        val = (float)(sqrt(2.0 / NMEL) * cos(M_PI * (double)N * ((double)K + 0.5) / NMEL));
    ws[OFF_DCT + idx] = (_Float16)val;
}

// ---------------- fused MFCC kernel: 16 frames x 1 channel per block ----------------
__global__ __launch_bounds__(128)
void mfcc_fused(const float* __restrict__ wav, const _Float16* __restrict__ ws,
                float* __restrict__ out) {
    __shared__ float s_samp[15 * STEP + WIN];                 // 5440 raw samples
    __shared__ __align__(32) _Float16 s_A  [KS1 * 32 * 16];   // windowed A frags (16x640)
    __shared__ __align__(32) _Float16 s_mag[KS2 * 32 * 16];   // |DFT| A frags (16x544)
    __shared__ __align__(32) _Float16 s_lm [KS3 * 32 * 16];   // log-mel A frags (16x64)

    const int tid   = threadIdx.x;
    const int lane  = tid & 31;
    const int wave  = tid >> 5;
    const int fbase = blockIdx.x * 16;
    const int ch    = blockIdx.y;
    const int s0    = fbase * STEP;

    // ---- 1) stage raw samples (50% frame overlap deduped) ----
    for (int s = tid; s < 15 * STEP + WIN; s += 128) {
        int g = s0 + s;
        if (g < SAMPLES) {
#if defined(__AMDGCN__) && __has_builtin(__builtin_amdgcn_global_load_async_to_lds_b32)
            typedef __attribute__((address_space(1))) int as1_int_t;
            typedef __attribute__((address_space(3))) int as3_int_t;
            __builtin_amdgcn_global_load_async_to_lds_b32(
                (as1_int_t*)(wav + (size_t)g * CHANNELS + ch),
                (as3_int_t*)&s_samp[s], 0, 0);
#else
            s_samp[s] = wav[(size_t)g * CHANNELS + ch];
#endif
        } else {
            s_samp[s] = 0.f;
        }
    }
#if defined(__AMDGCN__) && __has_builtin(__builtin_amdgcn_global_load_async_to_lds_b32)
#if __has_builtin(__builtin_amdgcn_s_wait_asynccnt)
    __builtin_amdgcn_s_wait_asynccnt(0);
#else
    asm volatile("s_wait_asynccnt 0" ::: "memory");
#endif
#endif
    __syncthreads();

    // ---- 2) Hann window -> f16 A fragments, zero log-mel pad ----
    for (int e = tid; e < KS1 * 32 * 16; e += 128) {
        int ks = e >> 9, ln = (e >> 4) & 31, j = e & 15;
        int n = ks * 32 + kpat(ln, j);
        int m = ln & 15;
        float h = 0.5f - 0.5f * __cosf(6.2831853071795864f * (float)n / (float)WIN);
        s_A[e] = (_Float16)(s_samp[m * STEP + n] * h);
    }
    for (int e = tid; e < KS3 * 32 * 16; e += 128) s_lm[e] = (_Float16)0.f;
    __syncthreads();

    // ---- 3) |rDFT| as two WMMA GEMM chains; scatter into stage-2 A layout ----
    for (int nt = wave; nt < NT1; nt += 4) {
        v8f cre = {}, cim = {};
        const _Float16* bc = ws + OFF_COS + (size_t)nt * KS1 * 512;
        const _Float16* bs = ws + OFF_SIN + (size_t)nt * KS1 * 512;
        for (int ks = 0; ks < KS1; ++ks) {
            if (ks + 1 < KS1) __builtin_prefetch(&bc[(ks + 1) * 512], 0, 0);
            v16h a  = *(const v16h*)&s_A[(ks * 32 + lane) * 16];
            v16h vb = *(const v16h*)&bc[(ks * 32 + lane) * 16];
            cre = __builtin_amdgcn_wmma_f32_16x16x32_f16(false, a, false, vb,
                                                         (short)0, cre, false, false);
            v16h vs = *(const v16h*)&bs[(ks * 32 + lane) * 16];
            cim = __builtin_amdgcn_wmma_f32_16x16x32_f16(false, a, false, vs,
                                                         (short)0, cim, false, false);
        }
        int K  = nt * 16 + (lane & 15);         // freq bin = stage-2 K index
        int k2 = K >> 5, r5 = K & 31;
        int jj = ((r5 >> 4) << 3) + (r5 & 7);
        int lh = (r5 >> 3) & 1;
        for (int r = 0; r < 8; ++r) {
            int mr = r + ((lane >> 4) << 3);    // frame row
            float mag = sqrtf(cre[r] * cre[r] + cim[r] * cim[r]);
            s_mag[(k2 * 32 + mr + (lh << 4)) * 16 + jj] = (_Float16)mag;
        }
    }
    __syncthreads();

    // ---- 4) mel projection (WMMA) + log; scatter into stage-3 A layout ----
    if (wave < NT2) {
        v8f acc = {};
        const _Float16* bm = ws + OFF_MEL + (size_t)wave * KS2 * 512;
        for (int ks = 0; ks < KS2; ++ks) {
            v16h a = *(const v16h*)&s_mag[(ks * 32 + lane) * 16];
            v16h b = *(const v16h*)&bm[(ks * 32 + lane) * 16];
            acc = __builtin_amdgcn_wmma_f32_16x16x32_f16(false, a, false, b,
                                                         (short)0, acc, false, false);
        }
        int K  = wave * 16 + (lane & 15);       // mel channel = stage-3 K index
        int k2 = K >> 5, r5 = K & 31;
        int jj = ((r5 >> 4) << 3) + (r5 & 7);
        int lh = (r5 >> 3) & 1;
        for (int r = 0; r < 8; ++r) {
            int mr = r + ((lane >> 4) << 3);
            float v = (K < NMEL) ? __logf(fmaxf(acc[r], 0.f) + 1e-12f) : 0.f;
            s_lm[(k2 * 32 + mr + (lh << 4)) * 16 + jj] = (_Float16)v;
        }
    }
    __syncthreads();

    // ---- 5) DCT (WMMA) + clip + store ----
    if (wave == 0) {
        v8f acc = {};
        const _Float16* bd = ws + OFF_DCT;
        for (int ks = 0; ks < KS3; ++ks) {
            v16h a = *(const v16h*)&s_lm[(ks * 32 + lane) * 16];
            v16h b = *(const v16h*)&bd[(ks * 32 + lane) * 16];
            acc = __builtin_amdgcn_wmma_f32_16x16x32_f16(false, a, false, b,
                                                         (short)0, acc, false, false);
        }
        int N = lane & 15;                      // mfcc coefficient
        for (int r = 0; r < 8; ++r) {
            int f = fbase + r + ((lane >> 4) << 3);
            if (N < NMFCC && f < NFRAMES) {
                float v = fminf(fmaxf(acc[r], -CLIPV), CLIPV);
                out[((size_t)ch * NFRAMES + f) * NMFCC + N] = v;
            }
        }
    }
}

extern "C" void kernel_launch(void* const* d_in, const int* in_sizes, int n_in,
                              void* d_out, int out_size, void* d_ws, size_t ws_size,
                              hipStream_t stream) {
    (void)in_sizes; (void)n_in; (void)out_size; (void)ws_size;
    const float* wav = (const float*)d_in[0];
    float* out = (float*)d_out;
    _Float16* ws = (_Float16*)d_ws;   // needs ~1.45 MB

    build_dft<<<(N_COS  + 255) / 256, 256, 0, stream>>>(ws);
    build_mel<<<(N_MELW + 255) / 256, 256, 0, stream>>>(ws);
    build_dct<<<(N_DCTW + 255) / 256, 256, 0, stream>>>(ws);

    dim3 grid(FBLOCKS, CHANNELS);
    mfcc_fused<<<grid, 128, 0, stream>>>(wav, ws, out);
}
